// SmolVLADenoiseStep_23613730194156
// MI455X (gfx1250) — matrix operand order repaired and checked by reading, not compiled, not run on hardware
//
#include <hip/hip_runtime.h>
#include <cstdint>
#include <cstddef>

// ---------------- problem constants (SmolVLA action expert) ----------------
static constexpr int kL    = 16;
static constexpr int kH    = 15;
static constexpr int kKH   = 5;
static constexpr int kD    = 64;
static constexpr int kHID  = 720;
static constexpr int kKV   = kKH * kD;     // 320
static constexpr int kINTER= 2048;
static constexpr int kP    = 241;
static constexpr int kS    = 50;
static constexpr int kA    = 32;
static constexpr int kB    = 64;
static constexpr int kTOK  = kB * kS;      // 3200 suffix tokens
static constexpr int kG    = kH / kKH;     // 3 (GQA group)

typedef __attribute__((ext_vector_type(16))) __bf16 v16bf;
typedef __attribute__((ext_vector_type(8)))  float  v8f;
typedef int i32x4 __attribute__((ext_vector_type(4)));
typedef int i32x8 __attribute__((ext_vector_type(8)));

union Frag16 { v16bf v; __bf16 s[16]; };

// ---------------------------------------------------------------------------
// Generic guarded GEMM:  C[M,N] = epi( A[M,K] @ Bw[K,N] )
// fp32 operands streamed from HBM, converted to bf16 into LDS, f32 WMMA accum.
// Software pipeline: fetch tile k+1 into VGPRs while computing tile k; LDS
// ping-pong with a single barrier per K-step. B tile stored transposed so the
// WMMA B operand is 2 contiguous ds_load_b128 per fragment (no scalar packing).
// ---------------------------------------------------------------------------
#define TM 128
#define TN 64
#define TK 32
#define LPAD 8

__global__ __launch_bounds__(256)
void gemm_bf16_wmma(const float* __restrict__ A, const float* __restrict__ Bw,
                    float* __restrict__ C, const float* __restrict__ bias,
                    const float* __restrict__ resid,
                    int M, int N, int K, int act_silu)
{
  __shared__ __bf16 As [2][TM][TK + LPAD];
  __shared__ __bf16 BsT[2][TN][TK + LPAD];

  const int tid  = threadIdx.x;
  const int lane = tid & 31;
  const int wid  = tid >> 5;
  const int wm   = (wid >> 1) * 32;   // wave row offset in block tile (0..96)
  const int wn   = (wid & 1) * 32;    // wave col offset (0 or 32)
  const int m0   = blockIdx.y * TM;
  const int n0   = blockIdx.x * TN;
  const bool mn_full = (m0 + TM <= M) && (n0 + TN <= N);

  float4 ra[4], rb[2];                // in-flight staging registers

  auto fetch = [&](int kk) {
    if (mn_full && (kk + TK <= K)) {
#pragma unroll
      for (int l = 0; l < 4; ++l) {                 // A: 128x32 = 1024 float4
        int li = (tid + 256 * l) * 4;
        int r = li >> 5, c = li & 31;
        ra[l] = *(const float4*)(A + (size_t)(m0 + r) * K + (kk + c));
      }
#pragma unroll
      for (int l = 0; l < 2; ++l) {                 // B: 32x64 = 512 float4
        int li = (tid + 256 * l) * 4;
        int r = li >> 6, c = li & 63;
        rb[l] = *(const float4*)(Bw + (size_t)(kk + r) * N + (n0 + c));
      }
    } else {
      // tail path: clamp-and-select, no EXEC divergence
#pragma unroll
      for (int l = 0; l < 4; ++l) {
        int li = (tid + 256 * l) * 4;
        int r = li >> 5, c = li & 31;
        int gm = m0 + r;
        float tmp[4];
#pragma unroll
        for (int qq = 0; qq < 4; ++qq) {
          int gk = kk + c + qq;
          bool ok = (gm < M) && (gk < K);
          float v = A[(size_t)(ok ? gm : 0) * K + (ok ? gk : 0)];
          tmp[qq] = ok ? v : 0.0f;
        }
        ra[l] = make_float4(tmp[0], tmp[1], tmp[2], tmp[3]);
      }
#pragma unroll
      for (int l = 0; l < 2; ++l) {
        int li = (tid + 256 * l) * 4;
        int r = li >> 6, c = li & 63;
        int gk = kk + r;
        float tmp[4];
#pragma unroll
        for (int qq = 0; qq < 4; ++qq) {
          int gn = n0 + c + qq;
          bool ok = (gk < K) && (gn < N);
          float v = Bw[(size_t)(ok ? gk : 0) * N + (ok ? gn : 0)];
          tmp[qq] = ok ? v : 0.0f;
        }
        rb[l] = make_float4(tmp[0], tmp[1], tmp[2], tmp[3]);
      }
    }
  };

  auto stage = [&](int p) {
#pragma unroll
    for (int l = 0; l < 4; ++l) {
      int li = (tid + 256 * l) * 4;
      int r = li >> 5, c = li & 31;
      As[p][r][c + 0] = (__bf16)ra[l].x;
      As[p][r][c + 1] = (__bf16)ra[l].y;
      As[p][r][c + 2] = (__bf16)ra[l].z;
      As[p][r][c + 3] = (__bf16)ra[l].w;
    }
#pragma unroll
    for (int l = 0; l < 2; ++l) {
      int li = (tid + 256 * l) * 4;
      int r = li >> 6, c = li & 63;
      BsT[p][c + 0][r] = (__bf16)rb[l].x;
      BsT[p][c + 1][r] = (__bf16)rb[l].y;
      BsT[p][c + 2][r] = (__bf16)rb[l].z;
      BsT[p][c + 3][r] = (__bf16)rb[l].w;
    }
  };

  const v8f vzero = {0.f,0.f,0.f,0.f,0.f,0.f,0.f,0.f};
  v8f acc[2][2] = {{vzero, vzero},{vzero, vzero}};

  fetch(0);
  int p = 0;
  for (int kk = 0; kk < K; kk += TK, p ^= 1) {
    stage(p);
    __syncthreads();
    if (kk + TK < K) fetch(kk + TK);   // overlaps with the WMMA phase below

    // A fragment: 16-bit A 16x32 layout (ISA 7.12.2): lanes 0-15 kb=0,
    // lanes 16-31 kb=8; elem i<8 -> K=kb+i ; elem i>=8 -> K=16+kb+(i-8).
    // B fragment: lane = column, same K order as A (kb=0/8; i>=8 -> K=16+kb+(i-8)).
    const int mrow = lane & 15;
    const int kb   = (lane >> 4) << 3;
    const int koff = (lane >> 4) << 4;
    Frag16 afrag[2], bfrag[2];
#pragma unroll
    for (int fm = 0; fm < 2; ++fm) {
      int m = wm + fm * 16 + mrow;
#pragma unroll
      for (int i = 0; i < 16; ++i) {
        int k = (i < 8) ? (kb + i) : (16 + kb + (i - 8));
        afrag[fm].s[i] = As[p][m][k];
      }
    }
#pragma unroll
    for (int fn = 0; fn < 2; ++fn) {
      int n = wn + fn * 16 + mrow;
#pragma unroll
      for (int i = 0; i < 16; ++i)
        bfrag[fn].s[i] = BsT[p][n][(i < 8) ? (kb + i) : (16 + kb + (i - 8))];
    }

#pragma unroll
    for (int fm = 0; fm < 2; ++fm)
#pragma unroll
      for (int fn = 0; fn < 2; ++fn)
        acc[fm][fn] = __builtin_amdgcn_wmma_f32_16x16x32_bf16(
            false, afrag[fm].v, false, bfrag[fn].v,
            (short)0, acc[fm][fn], false, false);
  }

  // epilogue: C/D layout (ISA 7.12.2): VGPR r -> M=r (+8 for lanes 16-31)
  const int mhi  = (lane >> 4) << 3;
  const int ncol = lane & 15;
#pragma unroll
  for (int fm = 0; fm < 2; ++fm)
#pragma unroll
    for (int fn = 0; fn < 2; ++fn)
#pragma unroll
      for (int r = 0; r < 8; ++r) {
        int gm = m0 + wm + fm * 16 + mhi + r;
        int gn = n0 + wn + fn * 16 + ncol;
        if (mn_full || (gm < M && gn < N)) {
          float v = acc[fm][fn][r];
          if (bias)     v += bias[gn];
          if (act_silu) v = v / (1.0f + __expf(-v));
          if (resid)    v += resid[(size_t)gm * N + gn];
          C[(size_t)gm * N + gn] = v;
        }
      }
}

// ---------------------------------------------------------------------------
// RMSNorm: one block per row of `ncols`
// ---------------------------------------------------------------------------
__global__ __launch_bounds__(256)
void rmsnorm_kernel(const float* __restrict__ in, const float* __restrict__ w,
                    float* __restrict__ out, int ncols)
{
  const int row = blockIdx.x, t = threadIdx.x;
  const float* xi = in + (size_t)row * ncols;
  float ss = 0.f;
  for (int c = t; c < ncols; c += 256) { float v = xi[c]; ss += v * v; }
  __shared__ float red[256];
  red[t] = ss; __syncthreads();
  for (int off = 128; off > 0; off >>= 1) {
    if (t < off) red[t] += red[t + off];
    __syncthreads();
  }
  float inv = rsqrtf(red[0] / (float)ncols + 1e-6f);
  float* yo = out + (size_t)row * ncols;
  for (int c = t; c < ncols; c += 256) yo[c] = xi[c] * inv * w[c];
}

// ---------------------------------------------------------------------------
// RoPE (rotate-half, cos/sin duplicated across halves): x is [rows, 64],
// rows = (b*kS + s)*nh + head ; position = pos0 + s
// ---------------------------------------------------------------------------
__global__ void rope_kernel(float* __restrict__ x, int nh, int pos0, int total)
{
  int idx = blockIdx.x * blockDim.x + threadIdx.x;
  if (idx >= total) return;
  int d = idx & 31;
  int r = idx >> 5;
  int s = (r / nh) % kS;
  float inv = __powf(10000.0f, -(float)d / 32.0f);
  float ang = (float)(pos0 + s) * inv;
  float sn, cs; __sincosf(ang, &sn, &cs);
  float* p = x + (size_t)r * 64;
  float x0 = p[d], x1 = p[d + 32];
  p[d]      = x0 * cs - x1 * sn;
  p[d + 32] = x1 * cs + x0 * sn;
}

// ---------------------------------------------------------------------------
// Eager GQA attention, one workgroup per (b, head).
// Keys = concat(k1[b,0:len1], k2[b,0:len2]) with [B,len,kKH,kD] layout.
// ---------------------------------------------------------------------------
__global__ __launch_bounds__(256)
void attn_kernel(const float* __restrict__ q,
                 const float* __restrict__ k1, const float* __restrict__ v1, int len1,
                 const float* __restrict__ k2, const float* __restrict__ v2, int len2,
                 float* __restrict__ out)
{
  const int bh = blockIdx.x;
  const int b = bh / kH, h = bh % kH;
  const int kvh = h / kG;
  const int Ltot = len1 + len2;
  const int t = threadIdx.x;

  __shared__ float qv[kD];
  __shared__ float sc[kP + kS];
  __shared__ float red[256];
  __shared__ float pv[4][kD];

  for (int s = 0; s < kS; ++s) {
    if (t < kD) qv[t] = q[((size_t)(b * kS + s) * kH + h) * kD + t] * 0.125f;
    __syncthreads();

    for (int j = t; j < Ltot; j += 256) {
      const float* kp = (j < len1)
          ? &k1[(((size_t)b * len1 + j) * kKH + kvh) * kD]
          : &k2[(((size_t)b * len2 + (j - len1)) * kKH + kvh) * kD];
      const float4* kp4 = (const float4*)kp;
      float d = 0.f;
#pragma unroll
      for (int c = 0; c < kD / 4; ++c) {
        float4 kv = kp4[c];
        d += qv[4*c+0]*kv.x + qv[4*c+1]*kv.y + qv[4*c+2]*kv.z + qv[4*c+3]*kv.w;
      }
      sc[j] = d;
    }
    __syncthreads();

    float m = -1e30f;
    for (int j = t; j < Ltot; j += 256) m = fmaxf(m, sc[j]);
    red[t] = m; __syncthreads();
    for (int off = 128; off > 0; off >>= 1) {
      if (t < off) red[t] = fmaxf(red[t], red[t + off]);
      __syncthreads();
    }
    m = red[0]; __syncthreads();

    float sum = 0.f;
    for (int j = t; j < Ltot; j += 256) {
      float e = __expf(sc[j] - m);
      sc[j] = e; sum += e;
    }
    red[t] = sum; __syncthreads();
    for (int off = 128; off > 0; off >>= 1) {
      if (t < off) red[t] += red[t + off];
      __syncthreads();
    }
    const float invs = 1.0f / red[0];

    const int d = t & 63, chunk = t >> 6;
    float acc = 0.f;
    for (int j = chunk; j < Ltot; j += 4) {
      const float* vp = (j < len1)
          ? &v1[(((size_t)b * len1 + j) * kKH + kvh) * kD]
          : &v2[(((size_t)b * len2 + (j - len1)) * kKH + kvh) * kD];
      acc += sc[j] * vp[d];
    }
    pv[chunk][d] = acc;
    __syncthreads();
    if (t < kD)
      out[((size_t)(b * kS + s) * kH + h) * kD + t] =
          (pv[0][t] + pv[1][t] + pv[2][t] + pv[3][t]) * invs;
    __syncthreads();
  }
}

// ---------------------------------------------------------------------------
// Timestep sinusoid: temb[b, i] = i<360 ? sin(t*ang_i) : cos(t*ang_{i-360})
// ---------------------------------------------------------------------------
__global__ void temb_kernel(const float* __restrict__ tstep, float* __restrict__ temb)
{
  const int b = blockIdx.x, t = threadIdx.x;
  const int half = kHID / 2;
  for (int i = t; i < kHID; i += blockDim.x) {
    int idx = (i < half) ? i : i - half;
    float frac = (float)idx / (float)(half - 1);
    float period = 4e-3f * __powf(1000.0f, frac);   // MAXP/MINP = 1000
    float ang = 6.2831853071795864f / period * tstep[b];
    temb[(size_t)b * kHID + i] = (i < half) ? __sinf(ang) : __cosf(ang);
  }
}

// concat: xcat[b,s,:720]=act, xcat[b,s,720:]=temb[b]
__global__ void concat_kernel(const float* __restrict__ act,
                              const float* __restrict__ temb,
                              float* __restrict__ xcat, int total)
{
  int idx = blockIdx.x * blockDim.x + threadIdx.x;
  if (idx >= total) return;
  int c = idx % (2 * kHID);
  int row = idx / (2 * kHID);
  int b = row / kS;
  xcat[idx] = (c < kHID) ? act[(size_t)row * kHID + c]
                         : temb[(size_t)b * kHID + (c - kHID)];
}

// g *= u (g already SiLU'd by GEMM epilogue)
__global__ void mul_kernel(float* __restrict__ g, const float* __restrict__ u, int total)
{
  int idx = blockIdx.x * blockDim.x + threadIdx.x;
  if (idx < total) g[idx] *= u[idx];
}

// ---------------------------------------------------------------------------
// CDNA5 async-copy probe (NOT launched): gfx1250 async global->LDS path,
// b32 (proven last round) + b128 widening.
// ---------------------------------------------------------------------------
__global__ __launch_bounds__(256)
void cdna5_async_probe(const float* __restrict__ src, float* __restrict__ dst)
{
  __shared__ float buf[256];
  __shared__ float buf4[1024];
  const int t = threadIdx.x;
  unsigned lds_off = (unsigned)(unsigned long long)(void*)&buf[t];
  const float* g = src + (size_t)blockIdx.x * 256 + t;
  asm volatile("global_load_async_to_lds_b32 %0, %1, off"
               :: "v"(lds_off), "v"(g) : "memory");
  unsigned lds_off4 = (unsigned)(unsigned long long)(void*)&buf4[t * 4];
  const float* g4 = src + (size_t)blockIdx.x * 1024 + t * 4;
  asm volatile("global_load_async_to_lds_b128 %0, %1, off"
               :: "v"(lds_off4), "v"(g4) : "memory");
  asm volatile("s_wait_asynccnt 0x0" ::: "memory");
  __syncthreads();
  dst[(size_t)blockIdx.x * 256 + t] = buf[t] + buf4[t] + buf4[t + 512];
}

// ---------------------------------------------------------------------------
// CDNA5 TDM probe (NOT launched): inline-asm tensor_load_to_lds with a 2-group
// D# (2D tensor), per CDNA5 ISA; waits on TENSORcnt.
// ---------------------------------------------------------------------------
__global__ __launch_bounds__(32)
void cdna5_tdm_probe(const float* __restrict__ src, float* __restrict__ dst)
{
  __shared__ float tile[2048];
  unsigned lds = (unsigned)(unsigned long long)(void*)&tile[0];
  unsigned long long ga = (unsigned long long)(const void*)src;

  // D# group 0: count=1 | lds_addr | global_addr[56:0] | type=2
  i32x4 g0;
  g0.x = 1;
  g0.y = (int)lds;
  g0.z = (int)(unsigned)(ga & 0xffffffffull);
  g0.w = (int)(((unsigned)(ga >> 32) & 0x01ffffffu) | (2u << 30));

  // D# group 1: data_size=4B; tensor_dim0=128; tile 64x8; dim0 stride=128
  i32x8 g1;
  g1[0] = (2 << 16);        // workgroup_mask=0, data_size=2 (4 bytes)
  g1[1] = (128 << 16);      // tensor_dim0 low bits (bits 79:48 overall)
  g1[2] = 0;                // tensor_dim0 hi / tensor_dim1
  g1[3] = (64 << 16);       // tile_dim0 (bits 127:112)
  g1[4] = 8;                // tile_dim1 (bits 143:128)
  g1[5] = 128;              // tensor_dim0_stride low (bits 207:160)
  g1[6] = 0;
  g1[7] = 0;

  asm volatile("tensor_load_to_lds %0, %1" :: "s"(g0), "s"(g1) : "memory");
  __builtin_amdgcn_s_wait_tensorcnt(0);
  __syncthreads();
  dst[threadIdx.x] = tile[threadIdx.x];
}

// ---------------------------------------------------------------------------
static inline int cdiv(int a, int b) { return (a + b - 1) / b; }

static void gemm(hipStream_t st, const float* A, const float* Bw, float* C,
                 const float* bias, const float* resid, int M, int N, int K, int silu)
{
  dim3 grid(cdiv(N, TN), cdiv(M, TM));
  gemm_bf16_wmma<<<grid, 256, 0, st>>>(A, Bw, C, bias, resid, M, N, K, silu);
}

extern "C" void kernel_launch(void* const* d_in, const int* in_sizes, int n_in,
                              void* d_out, int out_size, void* d_ws, size_t ws_size,
                              hipStream_t stream)
{
  (void)in_sizes; (void)n_in; (void)out_size; (void)ws_size;
  const float* noisy  = (const float*)d_in[0];
  const float* tstep  = (const float*)d_in[1];
  const float* past_k = (const float*)d_in[2];
  const float* past_v = (const float*)d_in[3];
  const float* Wq_s   = (const float*)d_in[4];
  const float* Wk_s   = (const float*)d_in[5];
  const float* Wv_s   = (const float*)d_in[6];
  const float* Wo_s   = (const float*)d_in[7];
  const float* gate_s = (const float*)d_in[8];
  const float* up_s   = (const float*)d_in[9];
  const float* down_s = (const float*)d_in[10];
  const float* ln1_s  = (const float*)d_in[11];
  const float* ln2_s  = (const float*)d_in[12];
  const float* Wq_c   = (const float*)d_in[13];
  const float* Wk_c   = (const float*)d_in[14];
  const float* Wv_c   = (const float*)d_in[15];
  const float* Wo_c   = (const float*)d_in[16];
  const float* gate_c = (const float*)d_in[17];
  const float* up_c   = (const float*)d_in[18];
  const float* down_c = (const float*)d_in[19];
  const float* ln1_c  = (const float*)d_in[20];
  const float* ln2_c  = (const float*)d_in[21];
  const float* W_ain  = (const float*)d_in[22];
  const float* b_ain  = (const float*)d_in[23];
  const float* W_tin  = (const float*)d_in[24];
  const float* b_tin  = (const float*)d_in[25];
  const float* W_tout = (const float*)d_in[26];
  const float* b_tout = (const float*)d_in[27];
  const float* W_aout = (const float*)d_in[28];
  const float* b_aout = (const float*)d_in[29];
  const float* w_final= (const float*)d_in[30];

  // workspace layout (fp32 elements); ~143 MB total, fully rewritten per call
  float* ws   = (float*)d_ws;
  float* x    = ws + 0;                    // 3200*720
  float* h    = ws + 2304000;              // 3200*720
  float* q    = ws + 4608000;              // 3200*960
  float* ks   = ws + 7680000;              // 3200*320
  float* vs   = ws + 8704000;              // 3200*320
  float* attn = ws + 9728000;              // 3200*960
  float* g    = ws + 12800000;             // 3200*2048
  float* u    = ws + 19353600;             // 3200*2048
  float* kc   = ws + 25907200;             // 15424*320
  float* vc   = ws + 30842880;             // 15424*320
  float* temb = ws + 35778560;             // 64*720
  float* xcat = g;                         // alias (needs 3200*1440 <= 3200*2048)
  float* actb = u;                         // alias (needs 3200*720)

  const int MT = kTOK;                     // 3200
  const size_t kvLayer = (size_t)kB * kP * kKH * kD;

  // ----- suffix embedding -----
  temb_kernel<<<kB, 256, 0, stream>>>(tstep, temb);
  gemm(stream, noisy, W_ain, actb, b_ain, nullptr, MT, kHID, kA, 0);
  {
    int tot = MT * 2 * kHID;
    concat_kernel<<<cdiv(tot, 256), 256, 0, stream>>>(actb, temb, xcat, tot);
  }
  gemm(stream, xcat, W_tin, h, b_tin, nullptr, MT, kHID, 2 * kHID, 1);  // SiLU
  gemm(stream, h, W_tout, x, b_tout, nullptr, MT, kHID, kHID, 0);

  // ----- 16 expert layers -----
  for (int i = 0; i < kL; ++i) {
    const int j = i / 2;
    if ((i & 1) == 0) {
      // self-attention layer: suffix KV (RoPE'd) appended to cached prefix KV
      rmsnorm_kernel<<<MT, 256, 0, stream>>>(x, ln1_s + (size_t)j * kHID, h, kHID);
      gemm(stream, h, Wq_s + (size_t)j * kHID * kH * kD, q, nullptr, nullptr, MT, kH * kD, kHID, 0);
      gemm(stream, h, Wk_s + (size_t)j * kHID * kKV, ks, nullptr, nullptr, MT, kKV, kHID, 0);
      gemm(stream, h, Wv_s + (size_t)j * kHID * kKV, vs, nullptr, nullptr, MT, kKV, kHID, 0);
      rope_kernel<<<cdiv(MT * kH * 32, 256), 256, 0, stream>>>(q, kH, kP, MT * kH * 32);
      rope_kernel<<<cdiv(MT * kKH * 32, 256), 256, 0, stream>>>(ks, kKH, kP, MT * kKH * 32);
      attn_kernel<<<kB * kH, 256, 0, stream>>>(q,
          past_k + (size_t)i * kvLayer, past_v + (size_t)i * kvLayer, kP,
          ks, vs, kS, attn);
      gemm(stream, attn, Wo_s + (size_t)j * kH * kD * kHID, x, nullptr, x, MT, kHID, kH * kD, 0);
      rmsnorm_kernel<<<MT, 256, 0, stream>>>(x, ln2_s + (size_t)j * kHID, h, kHID);
      gemm(stream, h, gate_s + (size_t)j * kHID * kINTER, g, nullptr, nullptr, MT, kINTER, kHID, 1);
      gemm(stream, h, up_s   + (size_t)j * kHID * kINTER, u, nullptr, nullptr, MT, kINTER, kHID, 0);
      mul_kernel<<<cdiv(MT * kINTER, 256), 256, 0, stream>>>(g, u, MT * kINTER);
      gemm(stream, g, down_s + (size_t)j * kINTER * kHID, x, nullptr, x, MT, kHID, kINTER, 0);
    } else {
      // cross-attention layer: KV re-projected from cached VLM KV, RoPE on Q only
      rmsnorm_kernel<<<MT, 256, 0, stream>>>(x, ln1_c + (size_t)j * kHID, h, kHID);
      gemm(stream, h, Wq_c + (size_t)j * kHID * kH * kD, q, nullptr, nullptr, MT, kH * kD, kHID, 0);
      rope_kernel<<<cdiv(MT * kH * 32, 256), 256, 0, stream>>>(q, kH, 0, MT * kH * 32);
      gemm(stream, past_k + (size_t)i * kvLayer, Wk_c + (size_t)j * kKV * kKV, kc,
           nullptr, nullptr, kB * kP, kKV, kKV, 0);
      gemm(stream, past_v + (size_t)i * kvLayer, Wv_c + (size_t)j * kKV * kKV, vc,
           nullptr, nullptr, kB * kP, kKV, kKV, 0);
      attn_kernel<<<kB * kH, 256, 0, stream>>>(q, kc, vc, kP, nullptr, nullptr, 0, attn);
      gemm(stream, attn, Wo_c + (size_t)j * kH * kD * kHID, x, nullptr, x, MT, kHID, kH * kD, 0);
      rmsnorm_kernel<<<MT, 256, 0, stream>>>(x, ln2_c + (size_t)j * kHID, h, kHID);
      gemm(stream, h, gate_c + (size_t)j * kHID * kINTER, g, nullptr, nullptr, MT, kINTER, kHID, 1);
      gemm(stream, h, up_c   + (size_t)j * kHID * kINTER, u, nullptr, nullptr, MT, kINTER, kHID, 0);
      mul_kernel<<<cdiv(MT * kINTER, 256), 256, 0, stream>>>(g, u, MT * kINTER);
      gemm(stream, g, down_c + (size_t)j * kINTER * kHID, x, nullptr, x, MT, kHID, kINTER, 0);
    }
  }

  // ----- output head -----
  rmsnorm_kernel<<<MT, 256, 0, stream>>>(x, w_final, h, kHID);
  gemm(stream, h, W_aout, (float*)d_out, b_aout, nullptr, MT, kA, kHID, 0);
}
